// VectorQuantizer4_34703335751959
// MI455X (gfx1250) — matrix operand order repaired and checked by reading, not compiled, not run on hardware
//
#include <hip/hip_runtime.h>

// ---------------------------------------------------------------------------
// VQ-VAE block for MI455X (gfx1250), wave32 + v_wmma_f32_16x16x32_bf16.
//   conv1 (3x3 256->256, implicit GEMM, 32 px/WG, B-frag reuse over 2 M-tiles)
//   LN (standalone, bandwidth-trivial)
//   argmin over 8192-codebook: 128 px/WG, codebook tiles staged in 64KB LDS
//   gather + deterministic loss, conv2 (3x3 256->256) -> NCHW f32
// ---------------------------------------------------------------------------

typedef __attribute__((ext_vector_type(16))) __bf16        v16bf;
typedef __attribute__((ext_vector_type(8)))  float         v8f;
typedef __attribute__((ext_vector_type(4)))  unsigned int  u32x4;

union Frag { v16bf v; u32x4 q[2]; };

__device__ __forceinline__ unsigned short f2bf(float f) {
  unsigned int u = __float_as_uint(f);
  u += 0x7FFFu + ((u >> 16) & 1u);           // round-to-nearest-even
  return (unsigned short)(u >> 16);
}

__device__ __forceinline__ v8f wmma_bf16(const Frag& a, const Frag& b, v8f c) {
  return __builtin_amdgcn_wmma_f32_16x16x32_bf16(
      false, a.v, false, b.v, (short)0, c, false, false);
}

constexpr int ED   = 256;       // E_DIM == IN_C
constexpr int NE   = 8192;      // codebook entries
constexpr int NPIX = 16384;     // 16*32*32
constexpr int KT   = 2304;      // 9 taps * 256
constexpr int OUTN = NPIX * ED; // 4194304

// ---------------------------------------------------------------------------
// Prep kernels
// ---------------------------------------------------------------------------

// z [16,256,32,32] f32 NCHW -> zin [pix][c] bf16 (NHWC)
__global__ __launch_bounds__(256) void k_prep_z(const float* __restrict__ z,
                                                unsigned short* __restrict__ zin) {
  int i = blockIdx.x * 256 + threadIdx.x;
  int c = i & 255, pix = i >> 8;
  int b = pix >> 10, yx = pix & 1023;
  zin[i] = f2bf(z[(((b << 8) + c) << 10) + yx]);
}

// w [O=256][I=256][3][3] f32 -> wt [O][k = tap*256 + i] bf16
__global__ __launch_bounds__(256) void k_prep_w(const float* __restrict__ w,
                                                unsigned short* __restrict__ wt) {
  int t = blockIdx.x * 256 + threadIdx.x;
  int o = t / KT, k = t - o * KT;
  int tap = k >> 8, i = k & 255;
  wt[t] = f2bf(w[(o * 256 + i) * 9 + tap]);
}

// codebook [8192][256] f32 -> bf16 copy + row norms
__global__ __launch_bounds__(256) void k_prep_cb(const float* __restrict__ cb,
                                                 unsigned short* __restrict__ cbh,
                                                 float* __restrict__ cnorm) {
  int n = blockIdx.x, t = threadIdx.x;
  float v = cb[n * ED + t];
  cbh[n * ED + t] = f2bf(v);
  float s = v * v;
  for (int off = 16; off; off >>= 1) s += __shfl_xor(s, off, 32);
  __shared__ float r[8];
  if ((t & 31) == 0) r[t >> 5] = s;
  __syncthreads();
  if (t == 0) {
    float tot = 0.f;
    for (int w = 0; w < 8; ++w) tot += r[w];
    cnorm[n] = tot;
  }
}

// ---------------------------------------------------------------------------
// Conv (implicit GEMM): 32 pixels (one image row) x 256 outputs per WG.
// 8 waves partition outputs (32 each); each wave: 2 M-tiles x 2 N-tiles,
// B fragments reused across both M-tiles. A from LDS window 3x34x256 bf16.
// CONV1: writes ze [pix][256] f32.  CONV2: writes out NCHW f32 (+bias only).
// ---------------------------------------------------------------------------
template <int MODE>   // 0 = conv1 (ze out), 1 = conv2 (NCHW out)
__global__ __launch_bounds__(256)
void k_conv(const unsigned short* __restrict__ src,   // [NPIX][256] bf16 NHWC
            const unsigned short* __restrict__ wt,    // [256][2304] bf16
            const float* __restrict__ bias,
            float* __restrict__ dst) {
  __shared__ __align__(16) unsigned short lx[3 * 34 * 256];   // 52224 B

  const int t = threadIdx.x;
  const int wave = t >> 5, lane = t & 31, hl = lane >> 4, m = lane & 15;
  const int wg = blockIdx.x;            // 512 WGs: b*32 + y
  const int b = wg >> 5, y = wg & 31;
  const int pix0 = wg << 5;             // (b*32+y)*32

  // stage 3x34x256 bf16 window (zero-padded borders), uint granularity
  {
    unsigned int* lxu = (unsigned int*)lx;
    const unsigned int* s = (const unsigned int*)src;
    for (int i = 0; i < 51; ++i) {
      int e = i * 256 + t;              // 0..13055 uints
      int cu = e & 127, rc = e >> 7;    // rc = ry*34+cx in 0..101
      int ry = rc / 34, cx = rc - ry * 34;
      int yy = y + ry - 1, xx = cx - 1;
      unsigned int val = 0u;
      if ((unsigned)yy < 32u && (unsigned)xx < 32u)
        val = s[(((b << 10) + (yy << 5) + xx) << 7) + cu];
      lxu[e] = val;
    }
  }
  __syncthreads();

  const int obase = wave << 5;
  const int o0 = obase + m, o1 = obase + 16 + m;
  v8f acc00 = {}, acc01 = {}, acc10 = {}, acc11 = {};  // [mt][nt]
  for (int ky = 0; ky < 3; ++ky)
    for (int kx = 0; kx < 3; ++kx) {
      const unsigned short* arow0 = &lx[(ky * 34 + m + kx) << 8];
      const unsigned short* arow1 = arow0 + (16 << 8);
      const int kglob = (ky * 3 + kx) << 8;
      const unsigned short* b0p = &wt[o0 * KT + kglob];
      const unsigned short* b1p = &wt[o1 * KT + kglob];
#pragma unroll
      for (int kc = 0; kc < 8; ++kc) {
        const int kbA = (kc << 5) + (hl << 3);   // A runs [8h,+8) & [16+8h,+8)
        const int kbB = (kc << 5) + (hl << 3);   // B runs [8h,+8) & [16+8h,+8)
        Frag a0, a1, fb0, fb1;
        a0.q[0]  = *(const u32x4*)(arow0 + kbA);
        a0.q[1]  = *(const u32x4*)(arow0 + kbA + 16);
        a1.q[0]  = *(const u32x4*)(arow1 + kbA);
        a1.q[1]  = *(const u32x4*)(arow1 + kbA + 16);
        fb0.q[0] = *(const u32x4*)(b0p + kbB);
        fb0.q[1] = *(const u32x4*)(b0p + kbB + 16);
        fb1.q[0] = *(const u32x4*)(b1p + kbB);
        fb1.q[1] = *(const u32x4*)(b1p + kbB + 16);
        acc00 = wmma_bf16(a0, fb0, acc00);
        acc01 = wmma_bf16(a0, fb1, acc01);
        acc10 = wmma_bf16(a1, fb0, acc10);
        acc11 = wmma_bf16(a1, fb1, acc11);
      }
    }
  const float bb0 = bias[o0], bb1 = bias[o1];
#pragma unroll
  for (int v = 0; v < 8; ++v) {
    int r0 = v + (hl << 3);        // pixel x within first M-tile
    int r1 = r0 + 16;
    if (MODE == 0) {
      dst[((pix0 + r0) << 8) + o0] = acc00[v] + bb0;
      dst[((pix0 + r0) << 8) + o1] = acc01[v] + bb1;
      dst[((pix0 + r1) << 8) + o0] = acc10[v] + bb0;
      dst[((pix0 + r1) << 8) + o1] = acc11[v] + bb1;
    } else {
      int base0 = (((b << 8) + o0) << 10) + (y << 5);
      int base1 = (((b << 8) + o1) << 10) + (y << 5);
      dst[base0 + r0] = acc00[v] + bb0;
      dst[base1 + r0] = acc01[v] + bb1;
      dst[base0 + r1] = acc10[v] + bb0;
      dst[base1 + r1] = acc11[v] + bb1;
    }
  }
}

// ---------------------------------------------------------------------------
// Standalone LayerNorm: one block per pixel (coalesced, bandwidth-trivial)
// ---------------------------------------------------------------------------
__global__ __launch_bounds__(256)
void k_ln(const float* __restrict__ ze, const float* __restrict__ g,
          const float* __restrict__ bb, unsigned short* __restrict__ zf,
          float* __restrict__ znorm) {
  int p = blockIdx.x, t = threadIdx.x;
  float v = ze[(p << 8) + t];
  float s = v, ss = v * v;
  for (int off = 16; off; off >>= 1) {
    s  += __shfl_xor(s, off, 32);
    ss += __shfl_xor(ss, off, 32);
  }
  __shared__ float rs_[8], rss_[8], bc[2];
  if ((t & 31) == 0) { rs_[t >> 5] = s; rss_[t >> 5] = ss; }
  __syncthreads();
  if (t == 0) {
    float a = 0.f, b2 = 0.f;
    for (int w = 0; w < 8; ++w) { a += rs_[w]; b2 += rss_[w]; }
    float mu = a * (1.f / 256.f);
    float var = b2 * (1.f / 256.f) - mu * mu;
    bc[0] = mu; bc[1] = rsqrtf(var + 1e-5f);
  }
  __syncthreads();
  float zfv = (v - bc[0]) * bc[1] * g[t] + bb[t];
  zf[(p << 8) + t] = f2bf(zfv);
  float zn = zfv * zfv;
  for (int off = 16; off; off >>= 1) zn += __shfl_xor(zn, off, 32);
  if ((t & 31) == 0) rs_[t >> 5] = zn;
  __syncthreads();
  if (t == 0) {
    float a = 0.f;
    for (int w = 0; w < 8; ++w) a += rs_[w];
    znorm[p] = a;
  }
}

// ---------------------------------------------------------------------------
// Codebook argmin: 128 pixels per WG; 8 waves own 16 rows each; 128-column
// codebook tiles staged in 64KB LDS shared by all waves (8x less L2 traffic).
// ---------------------------------------------------------------------------
__global__ __launch_bounds__(256)
void k_argmin(const unsigned short* __restrict__ zf, const float* __restrict__ znorm,
              const unsigned short* __restrict__ cb, const float* __restrict__ cnorm,
              int* __restrict__ idx_out) {
  __shared__ __align__(16) unsigned short lcb[128 * 256];     // 64 KB
  const int t = threadIdx.x, wave = t >> 5, lane = t & 31, hl = lane >> 4, m = lane & 15;
  const int pw0 = (blockIdx.x << 7) + (wave << 4);            // this wave's 16 rows

  Frag a[8];
#pragma unroll
  for (int kc = 0; kc < 8; ++kc) {
    const unsigned short* p = &zf[((pw0 + m) << 8) + (kc << 5) + (hl << 3)];
    a[kc].q[0] = *(const u32x4*)p;
    a[kc].q[1] = *(const u32x4*)(p + 16);
  }
  float zn[8];
#pragma unroll
  for (int v = 0; v < 8; ++v) zn[v] = znorm[pw0 + v + (hl << 3)];

  float bd[8]; int bn[8];
#pragma unroll
  for (int v = 0; v < 8; ++v) { bd[v] = 3.4e38f; bn[v] = 0; }

  unsigned int* lcbu = (unsigned int*)lcb;
  const unsigned int* cbu = (const unsigned int*)cb;

  for (int ct = 0; ct < 64; ++ct) {
    const int ncol0 = ct << 7;
    // cooperative stage: 128 codebook rows (64KB) into LDS, coalesced
    for (int i = 0; i < 64; ++i) {
      int e = i * 256 + t;                   // 0..16383 uints
      lcbu[e] = cbu[(ncol0 << 7) + e];
    }
    __syncthreads();
#pragma unroll 1
    for (int sub = 0; sub < 8; ++sub) {
      const int nloc = (sub << 4) + m;       // this lane's column in tile
      const unsigned short* bp = &lcb[nloc << 8];
      v8f acc = {};
#pragma unroll
      for (int kc = 0; kc < 8; ++kc) {
        const int kbB = (kc << 5) + (hl << 3);
        Frag fb;
        fb.q[0] = *(const u32x4*)(bp + kbB);
        fb.q[1] = *(const u32x4*)(bp + kbB + 16);
        acc = wmma_bf16(a[kc], fb, acc);
      }
      const int n = ncol0 + nloc;
      float cn = cnorm[n];
#pragma unroll
      for (int v = 0; v < 8; ++v) {
        float d = zn[v] + cn - 2.f * acc[v];
        if (d < bd[v] || (d == bd[v] && n < bn[v])) { bd[v] = d; bn[v] = n; }
      }
    }
    __syncthreads();
  }
  // reduce over the 16 lanes of each half (same 8 rows, different columns)
#pragma unroll
  for (int v = 0; v < 8; ++v) {
    for (int off = 8; off; off >>= 1) {
      float od = __shfl_xor(bd[v], off, 16);
      int   on = __shfl_xor(bn[v], off, 16);
      if (od < bd[v] || (od == bd[v] && on < bn[v])) { bd[v] = od; bn[v] = on; }
    }
    if (m == 0) idx_out[pw0 + v + (hl << 3)] = bn[v];
  }
}

// ---------------------------------------------------------------------------
// Gather z_q = codebook[idx], per-pixel loss partial, bf16 copy for conv2
// ---------------------------------------------------------------------------
__global__ __launch_bounds__(256)
void k_gather(const int* __restrict__ idx, const float* __restrict__ codebook,
              const float* __restrict__ ze, unsigned short* __restrict__ zq,
              float* __restrict__ partial) {
  int p = blockIdx.x, t = threadIdx.x;
  int id = idx[p];
  float qv = codebook[id * ED + t];
  zq[(p << 8) + t] = f2bf(qv);
  float d = qv - ze[(p << 8) + t];
  float s = d * d;
  for (int off = 16; off; off >>= 1) s += __shfl_xor(s, off, 32);
  __shared__ float r[8];
  if ((t & 31) == 0) r[t >> 5] = s;
  __syncthreads();
  if (t == 0) {
    float tot = 0.f;
    for (int w = 0; w < 8; ++w) tot += r[w];
    partial[p] = tot;
  }
}

// deterministic fixed-tree loss reduction (no float atomics)
__global__ __launch_bounds__(256)
void k_loss_final(const float* __restrict__ partial, float* __restrict__ loss) {
  int t = threadIdx.x;
  float s = 0.f;
  for (int i = t; i < NPIX; i += 256) s += partial[i];
  for (int off = 16; off; off >>= 1) s += __shfl_xor(s, off, 32);
  __shared__ float r[8];
  if ((t & 31) == 0) r[t >> 5] = s;
  __syncthreads();
  if (t == 0) {
    float tot = 0.f;
    for (int w = 0; w < 8; ++w) tot += r[w];
    *loss = tot * (1.25f / (float)OUTN);   // (1+BETA) * mean
  }
}

// ---------------------------------------------------------------------------
// Host launcher
// ---------------------------------------------------------------------------
extern "C" void kernel_launch(void* const* d_in, const int* in_sizes, int n_in,
                              void* d_out, int out_size, void* d_ws, size_t ws_size,
                              hipStream_t stream) {
  const float* z        = (const float*)d_in[0];
  const float* emb_w    = (const float*)d_in[1];
  const float* emb_b    = (const float*)d_in[2];
  const float* ln_g     = (const float*)d_in[3];
  const float* ln_b     = (const float*)d_in[4];
  const float* codebook = (const float*)d_in[5];
  const float* unemb_w  = (const float*)d_in[6];
  const float* unemb_b  = (const float*)d_in[7];

  char* ws = (char*)d_ws;
  float*          ze      = (float*)(ws + 0);                 // 16 MB
  unsigned short* zf      = (unsigned short*)(ws + 16777216); // 8 MB
  unsigned short* zin_zq  = (unsigned short*)(ws + 25165824); // 8 MB (zin, then zq)
  unsigned short* cbh     = (unsigned short*)(ws + 33554432); // 4 MB
  unsigned short* wemb    = (unsigned short*)(ws + 37748736); // 1.125 MB
  unsigned short* wun     = (unsigned short*)(ws + 38928384); // 1.125 MB
  float*          znorm   = (float*)(ws + 40108032);          // 64 KB
  float*          cnorm   = (float*)(ws + 40173568);          // 32 KB
  float*          partial = (float*)(ws + 40206336);          // 64 KB

  float* out  = (float*)d_out;
  float* loss = out + OUTN;
  int*   idxp = (int*)(out + OUTN + 1);

  k_prep_z <<<OUTN / 256, 256, 0, stream>>>(z, zin_zq);
  k_prep_w <<<(256 * KT) / 256, 256, 0, stream>>>(emb_w, wemb);
  k_prep_w <<<(256 * KT) / 256, 256, 0, stream>>>(unemb_w, wun);
  k_prep_cb<<<NE, 256, 0, stream>>>(codebook, cbh, cnorm);

  k_conv<0><<<NPIX / 32, 256, 0, stream>>>(zin_zq, wemb, emb_b, ze);
  k_ln     <<<NPIX, 256, 0, stream>>>(ze, ln_g, ln_b, zf, znorm);
  k_argmin <<<NPIX / 128, 256, 0, stream>>>(zf, znorm, cbh, cnorm, idxp);
  k_gather <<<NPIX, 256, 0, stream>>>(idxp, codebook, ze, zin_zq, partial);
  k_loss_final<<<1, 256, 0, stream>>>(partial, loss);
  k_conv<1><<<NPIX / 32, 256, 0, stream>>>(zin_zq, wun, unemb_b, out);
}
